// ScaleSpaceExtrema_57964878627378
// MI455X (gfx1250) — matrix operand order repaired and checked by reading, not compiled, not run on hardware
//
#include <hip/hip_runtime.h>

#define BB 8
#define SS 6
#define CC 128
#define OO 64
#define HH 128
#define WW 192
#define HW (HH * WW)            // 24576
#define TILES_PER_PLANE (HW/16) // 1536
#define NUM_TILES (BB * SS * TILES_PER_PLANE) // 73728
#define KPROP 100
#define CAP 16384
#define NEGV (-1e9f)

typedef __attribute__((ext_vector_type(16))) __bf16 v16bf;
typedef __attribute__((ext_vector_type(8)))  float  v8f;

// ---------------------------------------------------------------------------
// Kernel 1: fused 1x1-conv MLP head via bf16 WMMA, f32 accum.
// D(64 x 16px) = W1(64x128) * X(128x16px); epilogue: +b1, relu, dot w2, +b2, *sigma^2
// ---------------------------------------------------------------------------
__global__ void __launch_bounds__(256) resp_gemm_kernel(
    const float* __restrict__ feats,   // (B,S,C,H,W)
    const float* __restrict__ sigma,   // (S,)
    const float* __restrict__ w1,      // (64,128)
    const float* __restrict__ b1,      // (64,)
    const float* __restrict__ w2,      // (64,)
    const float* __restrict__ b2,      // (1,)
    float* __restrict__ resp)          // (B,S,H,W)
{
    // A fragments (bf16) staged in LDS in exact WMMA A-layout:
    // [kstep][otile][lane][slot], 32B per lane row -> 2x ds_load_b128 per fragment
    __shared__ __align__(32) __bf16 ldsA[4][4][32][16];
    // Epilogue constants staged in LDS:
    // [half][ot*8+r] = value at channel o = ot*16 + r + half*8
    __shared__ __align__(32) float lds_b1[2][32];
    __shared__ __align__(32) float lds_w2[2][32];

    const int tid  = threadIdx.x;
    const int lane = tid & 31;
    const int wave = tid >> 5;

    // Cooperative fill: convert w1 f32 -> bf16 into A-fragment layout.
    for (int idx = tid; idx < 4 * 4 * 32 * 16; idx += 256) {
        int j  = idx & 15;
        int l  = (idx >> 4) & 31;
        int ot = (idx >> 9) & 3;
        int ks = (idx >> 11) & 3;
        int r  = j >> 1;
        int kl = ((r < 4) ? (2 * r) : (16 + 2 * (r - 4))) + (j & 1) + ((l >= 16) ? 8 : 0);
        int o  = ot * 16 + (l & 15);
        int k  = ks * 32 + kl;
        ldsA[ks][ot][l][j] = (__bf16)w1[o * CC + k];
    }
    // Stage b1/w2 per-half layouts.
    if (tid < 64) {
        int h = tid >> 5;
        int e = tid & 31;
        int o = (e >> 3) * 16 + (e & 7) + h * 8;
        lds_b1[h][e] = b1[o];
        lds_w2[h][e] = w2[o];
    }
    __syncthreads();

    const int halfi = (lane >= 16) ? 1 : 0;
    const float b2s = b2[0];

    const int gwave  = blockIdx.x * 8 + wave;
    const int nwaves = gridDim.x * 8;
    const int n      = lane & 15;         // pixel within tile (B/D column)

    for (int tile = gwave; tile < NUM_TILES; tile += nwaves) {
        const int plane = tile / TILES_PER_PLANE;        // b*S + s
        const int p0    = (tile - plane * TILES_PER_PLANE) * 16;
        const int s     = plane % SS;
        const float* base = feats + (size_t)plane * CC * HW + p0;

        // Prefetch next tile's activations into cache hierarchy.
        int ntile = tile + nwaves;
        if (ntile < NUM_TILES) {
            int nplane = ntile / TILES_PER_PLANE;
            const float* nb = feats + (size_t)nplane * CC * HW
                            + (ntile - nplane * TILES_PER_PLANE) * 16;
            __builtin_prefetch(nb, 0, 1);
        }

        v8f acc[4] = {v8f{}, v8f{}, v8f{}, v8f{}};

#pragma unroll
        for (int ks = 0; ks < 4; ++ks) {
            // Build B fragment: k = 8*half + j (j<8), 16 + 8*half + (j-8) (j>=8), pixel n.
            // Features are streamed once -> non-temporal loads (don't pollute L2).
            v16bf bb;
#pragma unroll
            for (int j = 0; j < 16; ++j) {
                int kl = ((lane >= 16) ? 8 : 0) + ((j < 8) ? j : (16 + (j - 8)));
                float v = __builtin_nontemporal_load(&base[(size_t)(ks * 32 + kl) * HW + n]);
                bb[j] = (__bf16)v;
            }
#pragma unroll
            for (int ot = 0; ot < 4; ++ot) {
                // Opaque element offset: forces the A fragment to be re-read from
                // LDS each tile instead of being hoisted into 128 persistent VGPRs
                // (keeps wave under the 256-VGPR line -> higher occupancy).
                unsigned aoff = (unsigned)((((ks * 4 + ot) * 32) + lane) * 16);
                asm volatile("" : "+v"(aoff));
                v16bf aa = *(const v16bf*)&((const __bf16*)ldsA)[aoff];
                acc[ot] = __builtin_amdgcn_wmma_f32_16x16x32_bf16(
                    false, aa, false, bb, (short)0, acc[ot], false, false);
            }
        }

        // Epilogue: h = relu(D + b1); partial = sum_o h*w2 over this lane's 32 o's.
        float partial = 0.f;
#pragma unroll
        for (int ot = 0; ot < 4; ++ot) {
            unsigned eoff = (unsigned)(halfi * 32 + ot * 8);
            asm volatile("" : "+v"(eoff));
            v8f bv = *(const v8f*)&((const float*)lds_b1)[eoff];
            v8f wv = *(const v8f*)&((const float*)lds_w2)[eoff];
#pragma unroll
            for (int r = 0; r < 8; ++r) {
                float h = acc[ot][r] + bv[r];
                h = fmaxf(h, 0.f);
                partial = fmaf(h, wv[r], partial);
            }
        }

        // Lanes l and l^16 hold the same pixel -> one SWAPX16 swizzle-add.
        int pi = __float_as_int(partial);
#if __has_builtin(__builtin_amdgcn_ds_swizzle)
        int other = __builtin_amdgcn_ds_swizzle(pi, 0x401F); // xor 0x10, and 0x1f
#else
        int other = __shfl_xor(pi, 16, 32);
#endif
        float total = partial + __int_as_float(other);
        float sg  = sigma[s];
        float val = (total + b2s) * sg * sg;
        if (lane < 16)
            resp[(size_t)plane * HW + p0 + n] = val;   // RT store: stays in L2 for phase 2
    }
}

// ---------------------------------------------------------------------------
// Kernel 2: zero per-batch candidate counters (graph-safe, deterministic)
// ---------------------------------------------------------------------------
__global__ void zero_counts_kernel(int* __restrict__ counts) {
    if (threadIdx.x < BB) counts[threadIdx.x] = 0;
}

// ---------------------------------------------------------------------------
// Kernel 3: 3x3x3 scale-space local-max peak test + atomic compaction
// ---------------------------------------------------------------------------
__global__ void __launch_bounds__(256) peak_kernel(
    const float* __restrict__ resp,
    float* __restrict__ cand_score,
    int*   __restrict__ cand_idx,
    int*   __restrict__ counts)
{
    const int gid = blockIdx.x * blockDim.x + threadIdx.x;
    const int total = BB * SS * HW;
    if (gid >= total) return;

    const int b   = gid / (SS * HW);
    const int rem = gid - b * (SS * HW);
    const int s   = rem / HW;
    const int sp  = rem - s * HW;
    const int y   = sp / WW;
    const int x   = sp - y * WW;

    const float r0 = resp[gid];
    if (!(r0 > 0.f)) return;   // NEG padding can't beat r0>0, so OOB taps skip cleanly

    float lmax = NEGV;
    for (int ds = -1; ds <= 1; ++ds) {
        int ssb = s + ds;
        if (ssb < 0 || ssb >= SS) continue;
        for (int dy = -1; dy <= 1; ++dy) {
            int yy = y + dy;
            if (yy < 0 || yy >= HH) continue;
            const float* row = resp + (size_t)(b * SS + ssb) * HW + yy * WW;
            for (int dx = -1; dx <= 1; ++dx) {
                int xx = x + dx;
                if (xx < 0 || xx >= WW) continue;
                lmax = fmaxf(lmax, row[xx]);
            }
        }
    }
    if (fabsf(r0 - lmax) < 1e-6f) {
        int pos = atomicAdd(&counts[b], 1);
        if (pos < CAP) {
            cand_score[b * CAP + pos] = r0;
            cand_idx[b * CAP + pos]   = rem;   // flat index within batch: s*HW + sp
        }
    }
}

// ---------------------------------------------------------------------------
// Kernel 4: per-batch top-100 via iterative block argmax over candidates
// out layout (floats): scores[800] | coords[1600] | scales[800] | spatial[800]
// ---------------------------------------------------------------------------
__global__ void __launch_bounds__(256) topk_kernel(
    const float* __restrict__ sigma,
    float* __restrict__ cand_score,
    const int* __restrict__ cand_idx,
    const int* __restrict__ counts,
    float* __restrict__ out)
{
    __shared__ float s_sc[256];
    __shared__ int   s_fi[256];
    __shared__ int   s_pos[256];

    const int b   = blockIdx.x;
    const int tid = threadIdx.x;
    int cnt = counts[b];
    if (cnt > CAP) cnt = CAP;
    float* cs = cand_score + b * CAP;
    const int* ci = cand_idx + b * CAP;

    for (int k = 0; k < KPROP; ++k) {
        float bs = NEGV;
        int   bf = 0x7fffffff;
        int   bp = -1;
        for (int c = tid; c < cnt; c += 256) {
            float sc = cs[c];
            int   fi = ci[c];
            if (sc > bs || (sc == bs && fi < bf)) { bs = sc; bf = fi; bp = c; }
        }
        s_sc[tid] = bs; s_fi[tid] = bf; s_pos[tid] = bp;
        __syncthreads();
        for (int off = 128; off > 0; off >>= 1) {
            if (tid < off) {
                float os = s_sc[tid + off];
                int   of = s_fi[tid + off];
                if (os > s_sc[tid] || (os == s_sc[tid] && of < s_fi[tid])) {
                    s_sc[tid] = os; s_fi[tid] = of; s_pos[tid] = s_pos[tid + off];
                }
            }
            __syncthreads();
        }
        if (tid == 0) {
            float sc  = s_sc[0];
            int   fi  = s_fi[0];
            int   pos = s_pos[0];
            float score = (pos >= 0) ? fmaxf(sc, 0.f) : 0.f;
            int flat = (pos >= 0) ? fi : 0;
            int si = flat / HW;
            int sp = flat - si * HW;
            int y = sp / WW;
            int x = sp - y * WW;
            int ok = b * KPROP + k;
            out[ok] = score;
            out[800 + ok * 2 + 0] = (float)x / (float)WW;
            out[800 + ok * 2 + 1] = (float)y / (float)HH;
            int sic = si < 0 ? 0 : (si > SS - 1 ? SS - 1 : si);
            out[2400 + ok] = sigma[sic];
            out[3200 + ok] = (float)sp;
            if (pos >= 0) cs[pos] = NEGV;   // remove winner for next pass
        }
        __syncthreads();
    }
}

// ---------------------------------------------------------------------------
extern "C" void kernel_launch(void* const* d_in, const int* in_sizes, int n_in,
                              void* d_out, int out_size, void* d_ws, size_t ws_size,
                              hipStream_t stream) {
    const float* feats = (const float*)d_in[0];
    const float* sigma = (const float*)d_in[1];
    const float* w1    = (const float*)d_in[2];
    const float* b1    = (const float*)d_in[3];
    const float* w2    = (const float*)d_in[4];
    const float* b2    = (const float*)d_in[5];
    float* out = (float*)d_out;

    char* ws = (char*)d_ws;
    const size_t respBytes = (size_t)BB * SS * HW * sizeof(float); // 4.72 MB
    float* resp       = (float*)ws;
    int*   counts     = (int*)(ws + respBytes);
    float* cand_score = (float*)(ws + respBytes + 128);
    int*   cand_idx   = (int*)(ws + respBytes + 128 + (size_t)BB * CAP * sizeof(float));

    // Phase 1: WMMA response map (memory-bound: 603 MB feature read)
    resp_gemm_kernel<<<1536, 256, 0, stream>>>(feats, sigma, w1, b1, w2, b2, resp);

    // Phase 2: peaks + compaction
    zero_counts_kernel<<<1, 32, 0, stream>>>(counts);
    peak_kernel<<<(BB * SS * HW + 255) / 256, 256, 0, stream>>>(resp, cand_score, cand_idx, counts);

    // Phase 3: per-batch top-100 + output assembly
    topk_kernel<<<BB, 256, 0, stream>>>(sigma, cand_score, cand_idx, counts, out);
}
